// GroupedQueryAttention_69252052680926
// MI455X (gfx1250) — compile-verified, matrix operand order checked by reading, not run on hardware
//
#include <hip/hip_runtime.h>
#include <hip/hip_bf16.h>
#include <cstdint>

typedef _Float16 v16h __attribute__((ext_vector_type(16)));
typedef _Float16 v8h  __attribute__((ext_vector_type(8)));
typedef float    v8f  __attribute__((ext_vector_type(8)));
typedef float    v4f  __attribute__((ext_vector_type(4)));

#define B_   2
#define S_   2048
#define D_   2048
#define H_   16
#define KV_  4
#define HD_  128
#define REP_ 4
#define QKVW 3072   // 2048 (q) + 512 (k) + 512 (v)

static __device__ __forceinline__ v16h make_v16(v8h lo, v8h hi) {
    union { v16h v; v8h h[2]; } u;
    u.h[0] = lo; u.h[1] = hi;
    return u.v;
}

// ---- CDNA5 async global->LDS copy (ASYNCcnt-tracked, bypasses VGPRs) ------
static __device__ __forceinline__ void async_g2l_b128(uint32_t lds_byte_off,
                                                      const void* gaddr) {
    // LDS address is taken from the VDST VGPR; generic LDS pointer low 32 bits are the
    // wave-relative LDS offset (ISA 10.2 aperture mapping).
    asm volatile("global_load_async_to_lds_b128 %0, %1, off"
                 :: "v"(lds_byte_off), "v"(gaddr)
                 : "memory");
}
static __device__ __forceinline__ void wait_async0() {
#if __has_builtin(__builtin_amdgcn_s_wait_asynccnt)
    __builtin_amdgcn_s_wait_asynccnt(0);
#else
    asm volatile("s_wait_asynccnt 0x0" ::: "memory");
#endif
}
static __device__ __forceinline__ uint32_t lds_off(const void* p) {
    return (uint32_t)(uintptr_t)p;
}

// ---------------------------------------------------------------------------
// GEMM: C[M x N] = A[M x K] * W[N x K]^T   (weights row-major (N,K), fp32)
// AT = float (convert to f16 in LDS) or _Float16 (async-copied); OT = f16/f32.
// Block tile 128x128, K-step 32, 8 waves each computing 64x32 (4x2 WMMA tiles)
// ---------------------------------------------------------------------------
template <typename AT, typename OT>
__global__ void __launch_bounds__(256)
gemm_wmma(const AT* __restrict__ A, int lda,
          const float* __restrict__ W, int ldw,
          OT* __restrict__ C, int ldc, int K)
{
    constexpr int SA = 40;            // LDS row stride in halves (80B, 16B-aligned, conflict-free)
    __shared__ alignas(16) _Float16 sA[128 * SA];
    __shared__ alignas(16) _Float16 sW[128 * SA];

    const int tid  = threadIdx.x;
    const int lane = tid & 31;
    const int wid  = tid >> 5;
    const int wm   = wid >> 2;        // 0..1  (64 rows each)
    const int wn   = wid & 3;         // 0..3  (32 cols each)
    const int half = lane >> 4;       // 0/1
    const int l15  = lane & 15;

    const int rowBase = blockIdx.y * 128;
    const int colBase = blockIdx.x * 128;

    v8f vzero = {};
    v8f acc[4][2];
#pragma unroll
    for (int i = 0; i < 4; i++)
#pragma unroll
        for (int j = 0; j < 2; j++) acc[i][j] = vzero;

    const int r0 = tid >> 1;          // 0..127: row within tile
    const int h0 = tid & 1;           // which 16-wide half of the 32-wide k strip

    for (int kt = 0; kt < K; kt += 32) {
        // ---- stage A tile (128 x 32) -> LDS f16 ----
        {
            const AT* ap = A + (size_t)(rowBase + r0) * lda + kt + h0 * 16;
            _Float16* dst = &sA[r0 * SA + h0 * 16];
            __builtin_prefetch(ap + 32, 0, 1);   // next K strip -> global_prefetch_b8
            if constexpr (sizeof(AT) == 4) {
                v4f a0 = *(const v4f*)(ap + 0);
                v4f a1 = *(const v4f*)(ap + 4);
                v4f a2 = *(const v4f*)(ap + 8);
                v4f a3 = *(const v4f*)(ap + 12);
                v8h p0, p1;
#pragma unroll
                for (int i = 0; i < 4; i++) {
                    p0[i]     = (_Float16)a0[i];
                    p0[4 + i] = (_Float16)a1[i];
                    p1[i]     = (_Float16)a2[i];
                    p1[4 + i] = (_Float16)a3[i];
                }
                *(v8h*)(dst + 0) = p0;
                *(v8h*)(dst + 8) = p1;
            } else {
                // f16 A: pure copy -> CDNA5 async global->LDS DMA path
                async_g2l_b128(lds_off(dst + 0), ap + 0);
                async_g2l_b128(lds_off(dst + 8), ap + 8);
            }
        }
        // ---- stage W tile (128 x 32) fp32 -> LDS f16 ----
        {
            const float* wp = W + (size_t)(colBase + r0) * ldw + kt + h0 * 16;
            _Float16* dst = &sW[r0 * SA + h0 * 16];
            v4f a0 = *(const v4f*)(wp + 0);
            v4f a1 = *(const v4f*)(wp + 4);
            v4f a2 = *(const v4f*)(wp + 8);
            v4f a3 = *(const v4f*)(wp + 12);
            v8h p0, p1;
#pragma unroll
            for (int i = 0; i < 4; i++) {
                p0[i]     = (_Float16)a0[i];
                p0[4 + i] = (_Float16)a1[i];
                p1[i]     = (_Float16)a2[i];
                p1[4 + i] = (_Float16)a3[i];
            }
            *(v8h*)(dst + 0) = p0;
            *(v8h*)(dst + 8) = p1;
        }
        if constexpr (sizeof(AT) == 2) wait_async0();
        __syncthreads();

        // ---- fragments + WMMA ----
        v16h afrag[4];
#pragma unroll
        for (int mt = 0; mt < 4; mt++) {
            const _Float16* p = &sA[(wm * 64 + mt * 16 + l15) * SA + half * 8];
            afrag[mt] = make_v16(*(const v8h*)p, *(const v8h*)(p + 16));
        }
#pragma unroll
        for (int nt = 0; nt < 2; nt++) {
            const _Float16* p = &sW[(wn * 32 + nt * 16 + l15) * SA + half * 8];
            v16h bfrag = make_v16(*(const v8h*)p, *(const v8h*)(p + 16));
#pragma unroll
            for (int mt = 0; mt < 4; mt++)
                acc[mt][nt] = __builtin_amdgcn_wmma_f32_16x16x32_f16(
                    false, afrag[mt], false, bfrag, (short)0, acc[mt][nt], false, false);
        }
        __syncthreads();
    }

    // ---- epilogue: C layout row = r + half*8 (lanes), col = l15 ----
#pragma unroll
    for (int mt = 0; mt < 4; mt++)
#pragma unroll
        for (int nt = 0; nt < 2; nt++)
#pragma unroll
            for (int r = 0; r < 8; r++) {
                int row = rowBase + wm * 64 + mt * 16 + r + half * 8;
                int col = colBase + wn * 32 + nt * 16 + l15;
                C[(size_t)row * ldc + col] = (OT)acc[mt][nt][r];
            }
}

// ---------------------------------------------------------------------------
// RoPE in place on f16 QKV rows (q: 16 heads, k: 4 heads, 64 pairs each)
// ---------------------------------------------------------------------------
__global__ void __launch_bounds__(256)
rope_kernel(_Float16* __restrict__ qkv,
            const float* __restrict__ fcos,
            const float* __restrict__ fsin)
{
    const int row = blockIdx.x;            // b*S + s
    const int s   = row & (S_ - 1);
    _Float16* base = qkv + (size_t)row * QKVW;
    const float* cr = fcos + (size_t)s * (HD_ / 2);
    const float* sr = fsin + (size_t)s * (HD_ / 2);
    const int t = threadIdx.x;

    for (int p = t; p < H_ * (HD_ / 2); p += 256) {      // q part
        int hh = p >> 6, i = p & 63;
        _Float16* e = base + hh * HD_ + 2 * i;
        float t0 = (float)e[0], t1 = (float)e[1];
        float c = cr[i], sn = sr[i];
        e[0] = (_Float16)(t0 * c - t1 * sn);
        e[1] = (_Float16)(t0 * sn + t1 * c);
    }
    for (int p = t; p < KV_ * (HD_ / 2); p += 256) {     // k part
        int hh = p >> 6, i = p & 63;
        _Float16* e = base + D_ + hh * HD_ + 2 * i;
        float t0 = (float)e[0], t1 = (float)e[1];
        float c = cr[i], sn = sr[i];
        e[0] = (_Float16)(t0 * c - t1 * sn);
        e[1] = (_Float16)(t0 * sn + t1 * c);
    }
}

// ---------------------------------------------------------------------------
// Flash attention: block = 4 waves, 64 query rows per block, 32-key steps.
// grid = (S/64, B*H). Causal. Online softmax with half-wave shuffles.
// K tile staged via async global->LDS; V staged transposed (needs VALU pass).
// ---------------------------------------------------------------------------
__global__ void __launch_bounds__(128)
flash_attn(const _Float16* __restrict__ qkv, _Float16* __restrict__ attn)
{
    constexpr int SK = 136;                 // K LDS stride (halves): 272B, 16B-aligned
    constexpr int SV = 40;                  // Vt LDS stride
    constexpr int SP = 40;                  // P LDS stride
    __shared__ alignas(16) _Float16 sK[32 * SK];        // K tile, row-major (key, d)
    __shared__ alignas(16) _Float16 sV[HD_ * SV];       // V tile transposed (d, key)
    __shared__ alignas(16) _Float16 sP[4 * 16 * SP];    // per-wave P scratch

    const int tid  = threadIdx.x;
    const int lane = tid & 31;
    const int w    = tid >> 5;              // wave 0..3
    const int half = lane >> 4;
    const int l15  = lane & 15;

    const int qbase = blockIdx.x * 64;
    const int bh = blockIdx.y;
    const int b  = bh / H_;
    const int h  = bh % H_;
    const int kvh = h / REP_;

    const _Float16* Qbase = qkv + (size_t)(b * S_) * QKVW + h * HD_;
    const _Float16* Kbase = qkv + (size_t)(b * S_) * QKVW + D_ + kvh * HD_;
    const _Float16* Vbase = qkv + (size_t)(b * S_) * QKVW + D_ + KV_ * HD_ + kvh * HD_;

    // Q fragments: rows qbase + w*16 + l15, 4 d-steps of 32, held in registers
    v16h qf[4];
    {
        const _Float16* qp = Qbase + (size_t)(qbase + w * 16 + l15) * QKVW;
#pragma unroll
        for (int d = 0; d < 4; d++) {
            const _Float16* p = qp + d * 32 + half * 8;
            qf[d] = make_v16(*(const v8h*)p, *(const v8h*)(p + 16));
        }
    }

    v8f vzero = {};
    v8f o[8];
#pragma unroll
    for (int t = 0; t < 8; t++) o[t] = vzero;
    float m[8], l[8];
#pragma unroll
    for (int r = 0; r < 8; r++) { m[r] = -__builtin_inff(); l[r] = 0.f; }

    const float scale = 0.08838834764831845f;  // 1/sqrt(128)
    const int nsteps = (qbase + 64) / 32;

    for (int ks = 0; ks < nsteps; ks++) {
        const int kb = ks * 32;
        // ---- stage K (async DMA, row-major) and V (transposed) into LDS ----
        {
            const int key = tid >> 2, q4 = tid & 3;
            const _Float16* kp = Kbase + (size_t)(kb + key) * QKVW + q4 * 32;
            uint32_t kdst = lds_off(&sK[key * SK + q4 * 32]);
            async_g2l_b128(kdst + 0,  kp + 0);
            async_g2l_b128(kdst + 16, kp + 8);
            async_g2l_b128(kdst + 32, kp + 16);
            async_g2l_b128(kdst + 48, kp + 24);

            const _Float16* vp = Vbase + (size_t)(kb + key) * QKVW + q4 * 32;
#pragma unroll
            for (int c = 0; c < 4; c++) {
                v8h vv = *(const v8h*)(vp + c * 8);
#pragma unroll
                for (int i = 0; i < 8; i++)
                    sV[(q4 * 32 + c * 8 + i) * SV + key] = vv[i];
            }
        }
        wait_async0();
        __syncthreads();

        // ---- scores: S = Q * K^T, 2 tiles of 16 keys ----
        v8f sc[2];
#pragma unroll
        for (int nt = 0; nt < 2; nt++) {
            v8f s = vzero;
#pragma unroll
            for (int d = 0; d < 4; d++) {
                const _Float16* p = &sK[(nt * 16 + l15) * SK + d * 32 + half * 8];
                v16h kf = make_v16(*(const v8h*)p, *(const v8h*)(p + 16));
                s = __builtin_amdgcn_wmma_f32_16x16x32_f16(
                        false, qf[d], false, kf, (short)0, s, false, false);
            }
            sc[nt] = s;
        }

        // ---- scale + causal mask ----
#pragma unroll
        for (int nt = 0; nt < 2; nt++) {
            int kk = kb + nt * 16 + l15;
#pragma unroll
            for (int r = 0; r < 8; r++) {
                int qi = qbase + w * 16 + r + half * 8;
                float v = sc[nt][r] * scale;
                sc[nt][r] = (kk <= qi) ? v : -__builtin_inff();
            }
        }

        // ---- online softmax (row reductions across 16-lane halves) ----
        float fs[8];
#pragma unroll
        for (int r = 0; r < 8; r++) {
            float mx = fmaxf(sc[0][r], sc[1][r]);
#pragma unroll
            for (int d = 1; d < 16; d <<= 1) mx = fmaxf(mx, __shfl_xor(mx, d, 32));
            float mn = fmaxf(m[r], mx);
            float f  = __expf(m[r] - mn);
            float sum = 0.f;
#pragma unroll
            for (int nt = 0; nt < 2; nt++) {
                float e = __expf(sc[nt][r] - mn);
                sc[nt][r] = e;
                sum += e;
            }
#pragma unroll
            for (int d = 1; d < 16; d <<= 1) sum += __shfl_xor(sum, d, 32);
            l[r] = l[r] * f + sum;
            m[r] = mn;
            fs[r] = f;
        }
#pragma unroll
        for (int t = 0; t < 8; t++)
#pragma unroll
            for (int r = 0; r < 8; r++) o[t][r] *= fs[r];

        // ---- P: C-layout -> LDS -> A-layout (per-wave region, in-order DS) ----
        _Float16* pw = &sP[w * 16 * SP];
#pragma unroll
        for (int nt = 0; nt < 2; nt++)
#pragma unroll
            for (int r = 0; r < 8; r++)
                pw[(r + half * 8) * SP + nt * 16 + l15] = (_Float16)sc[nt][r];
        __threadfence_block();
        v16h pf;
        {
            const _Float16* p = pw + l15 * SP + half * 8;
            pf = make_v16(*(const v8h*)p, *(const v8h*)(p + 16));
        }

        // ---- O += P * V ----
#pragma unroll
        for (int t = 0; t < 8; t++) {
            const _Float16* p = &sV[(t * 16 + l15) * SV + half * 8];
            v16h vf = make_v16(*(const v8h*)p, *(const v8h*)(p + 16));
            o[t] = __builtin_amdgcn_wmma_f32_16x16x32_f16(
                       false, pf, false, vf, (short)0, o[t], false, false);
        }
        __syncthreads();
    }

    // ---- normalize + store f16 attn output (b, s, h, d) ----
#pragma unroll
    for (int r = 0; r < 8; r++) {
        float inv = 1.0f / l[r];
        int qi = qbase + w * 16 + r + half * 8;
        _Float16* op = attn + (size_t)(b * S_ + qi) * D_ + h * HD_;
#pragma unroll
        for (int t = 0; t < 8; t++)
            op[t * 16 + l15] = (_Float16)(o[t][r] * inv);
    }
}

// ---------------------------------------------------------------------------
extern "C" void kernel_launch(void* const* d_in, const int* in_sizes, int n_in,
                              void* d_out, int out_size, void* d_ws, size_t ws_size,
                              hipStream_t stream)
{
    (void)in_sizes; (void)n_in; (void)out_size; (void)ws_size;
    const float* x  = (const float*)d_in[0];
    const float* wq = (const float*)d_in[1];
    const float* wk = (const float*)d_in[2];
    const float* wv = (const float*)d_in[3];
    const float* wo = (const float*)d_in[4];
    const float* fc = (const float*)d_in[5];
    const float* fs = (const float*)d_in[6];
    float* out = (float*)d_out;

    _Float16* qkv  = (_Float16*)d_ws;                       // (B*S) x 3072 f16
    _Float16* attn = qkv + (size_t)(B_ * S_) * QKVW;        // (B*S) x 2048 f16

    const int M = B_ * S_;                                  // 4096
    dim3 blk(256);

    // QKV projections (f16 WMMA, f32 accumulate)
    gemm_wmma<float, _Float16><<<dim3(D_ / 128, M / 128), blk, 0, stream>>>(
        x, D_, wq, D_, qkv, QKVW, D_);
    gemm_wmma<float, _Float16><<<dim3((KV_ * HD_) / 128, M / 128), blk, 0, stream>>>(
        x, D_, wk, D_, qkv + D_, QKVW, D_);
    gemm_wmma<float, _Float16><<<dim3((KV_ * HD_) / 128, M / 128), blk, 0, stream>>>(
        x, D_, wv, D_, qkv + D_ + KV_ * HD_, QKVW, D_);

    // RoPE in place on q/k portions
    rope_kernel<<<dim3(M), blk, 0, stream>>>(qkv, fc, fs);

    // Flash attention
    flash_attn<<<dim3(S_ / 64, B_ * H_), dim3(128), 0, stream>>>(qkv, attn);

    // Output projection, f32 result
    gemm_wmma<_Float16, float><<<dim3(D_ / 128, M / 128), blk, 0, stream>>>(
        attn, D_, wo, D_, out, D_, D_);
}
